// SceneEngine_29695403884897
// MI455X (gfx1250) — matrix operand order repaired and checked by reading, not compiled, not run on hardware
//
#include <hip/hip_runtime.h>
#include <hip/hip_bf16.h>

// ---------------------------------------------------------------------------
// SceneEngine on gfx1250 (MI455X).
//   B=256 batches, K=16 panels, S=9 slots, P=511 occupancy patterns.
//   One block per batch element b: 512 threads = 16 wave32s.
//   Stage 1: position_logprob via v_wmma_f32_16x16x32_f16
//            (16 panels x 512 patterns, inner dim 18 padded to 32).
//   Stage 2: per-panel wave: exp/normalize/clip, number bins,
//            attribute subset-product sums, norms, rule mins.
// ---------------------------------------------------------------------------

typedef _Float16 v16h __attribute__((ext_vector_type(16)));
typedef float    v8f  __attribute__((ext_vector_type(8)));

#define NPAN 16
#define NSLOT 9
#define NPAT 512              // masks 1..511 valid; column 0 is dead
#define KDIM 18               // 9 slots x 2 exist states

// output float offsets (flat concat of reference tuple)
#define O_EXIST   0           // 256*16*9*2 = 73728
#define O_NUMBER  73728       // 256*16*9   = 36864
#define O_TYPE    110592      // 256*16*6   = 24576
#define O_NTYPE   135168      // 256*8*5    = 10240
#define O_RTYPE   145408      // 256
#define O_SIZE    145664      // 256*16*7   = 28672
#define O_NSIZE   174336      // 256*8*6    = 12288
#define O_RSIZE   186624      // 256
#define O_COLOR   186880      // 256*16*11  = 45056
#define O_NCOLOR  231936      // 256*8*10   = 20480
#define O_RCOLOR  252416      // 256

__global__ __launch_bounds__(512)
void scene_engine_kernel(const float* __restrict__ ex,   // (B,16,9,2)
                         const float* __restrict__ ty,   // (B,16,9,5)
                         const float* __restrict__ sz,   // (B,16,9,6)
                         const float* __restrict__ co,   // (B,16,9,10)
                         float* __restrict__ out) {
    __shared__ float s_exist[NPAN * KDIM];        // 288 floats
    __shared__ float s_logp[NPAN * NPAT];         // 16x512 = 32 KB
    __shared__ float s_ea[NPAN * 200];            // exp(attr): T@0(45) S@48(54) C@104(90)
    __shared__ float s_noninc[3 * NPAN];

    const int b    = blockIdx.x;
    const int tid  = threadIdx.x;
    const int lane = tid & 31;
    const int wv   = tid >> 5;        // wave id == panel id in stage 2
    const int half = lane >> 4;       // 0: lanes 0-15, 1: lanes 16-31
    const int l15  = lane & 15;

    // ---- Stage A: stage exist logprobs in LDS, emit exist_prob = exp ----
    if (tid < NPAN * KDIM) {
        float v = ex[b * (NPAN * KDIM) + tid];
        s_exist[tid] = v;
        out[O_EXIST + b * (NPAN * KDIM) + tid] = __expf(v);
    }

    // ---- Stage B2 prep: exp(attr logprobs) for this wave's panel ----
    {
        const int base = (b * NPAN + wv) * NSLOT;
        for (int i = lane; i < 189; i += 32) {
            float val; int dst;
            if (i < 45)        { int t = i;      int s = t / 5;  int d = t % 5;  val = ty[(base + s) * 5  + d]; dst = t; }
            else if (i < 99)   { int t = i - 45; int s = t / 6;  int d = t % 6;  val = sz[(base + s) * 6  + d]; dst = 48 + t; }
            else               { int t = i - 99; int s = t / 10; int d = t % 10; val = co[(base + s) * 10 + d]; dst = 104 + t; }
            s_ea[wv * 200 + dst] = __expf(val);
        }
    }
    __syncthreads();

    // ---- Stage B: WMMA position_logprob. A = exist (16 panels x 18),
    //      B = pattern one-hot (18 x 512), generated in registers. ----
    v16h afrag;
#pragma unroll
    for (int e = 0; e < 16; ++e) {
        // lanes 0-15: elems 0-7 -> K=e, elems 8-15 -> K=e+8 .. (K=16..23)
        // lanes 16-31: +8 (K=8..15 / 24..31)
        const int K = e + ((e >= 8) ? 8 : 0) + half * 8;
        afrag[e] = (K < KDIM) ? (_Float16)s_exist[l15 * KDIM + K] : (_Float16)0.0f;
    }
#pragma unroll
    for (int t = 0; t < 2; ++t) {
        const int nb = wv * 32 + t * 16;      // 16-column tile base; waves cover 512 cols
        const int m  = nb + l15;              // pattern mask = column index
        v16h bfrag;
#pragma unroll
        for (int e = 0; e < 16; ++e) {
            const int Kc = e + ((e >= 8) ? 8 : 0) + half * 8;     // lanes 0-15: K=0..7,16..23, lanes 16-31: K=8..15,24..31
            const int s  = Kc >> 1;
            const int sh = 8 - s;             // slot s occupies bit (8-s) of the mask
            const unsigned bit = (sh >= 0) ? ((unsigned(m) >> sh) & 1u) : 0u;
            const float val = ((Kc < KDIM) && (bit == unsigned(Kc & 1))) ? 1.0f : 0.0f;
            bfrag[e] = (_Float16)val;
        }
        v8f cacc = {};
        v8f dacc = __builtin_amdgcn_wmma_f32_16x16x32_f16(
            /*neg_a=*/false, afrag, /*neg_b=*/false, bfrag,
            /*c_mod=*/(short)0, cacc, /*reuse_a=*/false, /*reuse_b=*/false);
#pragma unroll
        for (int r = 0; r < 8; ++r)           // D: vgpr r, lanes<16 -> row r, lanes>=16 -> row r+8
            s_logp[(r + half * 8) * NPAT + nb + l15] = dacc[r];
    }
    __syncthreads();

    // ---- Stage C: per-panel wave processing (wave wv <-> panel k) ----
    const int k = wv;
    float ev[16];
    float psum = 0.0f;
#pragma unroll
    for (int j = 0; j < 16; ++j) {
        const int m = j * 32 + lane;          // conflict-free LDS reads
        float e = __expf(s_logp[k * NPAT + m]);
        if (m == 0) e = 0.0f;                 // mask 0 (no slots) is not a pattern
        ev[j] = e;
        psum += e;
    }
#pragma unroll
    for (int off = 16; off; off >>= 1) psum += __shfl_xor(psum, off, 32);
    const float inv = 1.0f / psum;

    // number_prob bins by popcount-1
    float bin[NSLOT] = {0, 0, 0, 0, 0, 0, 0, 0, 0};
#pragma unroll
    for (int j = 0; j < 16; ++j) {
        const int m = j * 32 + lane;
        const float pv = ev[j] * inv;
        const int c = __popc((unsigned)m) - 1; // m==0 -> -1, matches no bin
#pragma unroll
        for (int n = 0; n < NSLOT; ++n) bin[n] += (c == n) ? pv : 0.0f;
    }
#pragma unroll
    for (int n = 0; n < NSLOT; ++n) {
#pragma unroll
        for (int off = 16; off; off >>= 1) bin[n] += __shfl_xor(bin[n], off, 32);
    }

    // attribute subset-product sums
    float accT[5]  = {0, 0, 0, 0, 0};
    float accS[6]  = {0, 0, 0, 0, 0, 0};
    float accC[10] = {0, 0, 0, 0, 0, 0, 0, 0, 0, 0};
    const float* ea = &s_ea[wv * 200];
#pragma unroll
    for (int j = 0; j < 16; ++j) {
        const int m = j * 32 + lane;
        if (m) {
            const float pc = fmaxf(ev[j] * inv, 1e-10f);   // clip(position_prob, EPS)
            float vT[5], vS[6], vC[10];
#pragma unroll
            for (int d = 0; d < 5; ++d)  vT[d] = pc;
#pragma unroll
            for (int d = 0; d < 6; ++d)  vS[d] = pc;
#pragma unroll
            for (int d = 0; d < 10; ++d) vC[d] = pc;
            unsigned mm = (unsigned)m;
            while (mm) {
                const int bp = __ffs(mm) - 1;
                mm &= mm - 1;
                const int s = 8 - bp;                      // bit (8-s) <-> slot s
#pragma unroll
                for (int d = 0; d < 5; ++d)  vT[d] *= ea[s * 5 + d];
#pragma unroll
                for (int d = 0; d < 6; ++d)  vS[d] *= ea[48 + s * 6 + d];
#pragma unroll
                for (int d = 0; d < 10; ++d) vC[d] *= ea[104 + s * 10 + d];
            }
#pragma unroll
            for (int d = 0; d < 5; ++d)  accT[d] += vT[d];
#pragma unroll
            for (int d = 0; d < 6; ++d)  accS[d] += vS[d];
#pragma unroll
            for (int d = 0; d < 10; ++d) accC[d] += vC[d];
        }
    }
#pragma unroll
    for (int d = 0; d < 5; ++d) {
#pragma unroll
        for (int off = 16; off; off >>= 1) accT[d] += __shfl_xor(accT[d], off, 32);
    }
#pragma unroll
    for (int d = 0; d < 6; ++d) {
#pragma unroll
        for (int off = 16; off; off >>= 1) accS[d] += __shfl_xor(accS[d], off, 32);
    }
#pragma unroll
    for (int d = 0; d < 10; ++d) {
#pragma unroll
        for (int off = 16; off; off >>= 1) accC[d] += __shfl_xor(accC[d], off, 32);
    }

    if (lane == 0) {
        const int bk = b * NPAN + k;
        float sumT = 0.f, sumS = 0.f, sumC = 0.f;
#pragma unroll
        for (int d = 0; d < 5; ++d)  sumT += accT[d];
#pragma unroll
        for (int d = 0; d < 6; ++d)  sumS += accS[d];
#pragma unroll
        for (int d = 0; d < 10; ++d) sumC += accC[d];
        const float nT = fminf(sumT, 1.0f);
        const float nS = fminf(sumS, 1.0f);
        const float nC = fminf(sumC, 1.0f);

#pragma unroll
        for (int n = 0; n < NSLOT; ++n) out[O_NUMBER + bk * NSLOT + n] = bin[n];

#pragma unroll
        for (int d = 0; d < 5; ++d)  out[O_TYPE + bk * 6 + d] = accT[d];
        out[O_TYPE + bk * 6 + 5] = 1.0f - nT;
#pragma unroll
        for (int d = 0; d < 6; ++d)  out[O_SIZE + bk * 7 + d] = accS[d];
        out[O_SIZE + bk * 7 + 6] = 1.0f - nS;
#pragma unroll
        for (int d = 0; d < 10; ++d) out[O_COLOR + bk * 11 + d] = accC[d];
        out[O_COLOR + bk * 11 + 10] = 1.0f - nC;

        if (k < 8) {
            const int bk8 = b * 8 + k;
            const float iT = 1.0f / sumT, iS = 1.0f / sumS, iC = 1.0f / sumC;
#pragma unroll
            for (int d = 0; d < 5; ++d)  out[O_NTYPE + bk8 * 5 + d]  = accT[d] * iT;
#pragma unroll
            for (int d = 0; d < 6; ++d)  out[O_NSIZE + bk8 * 6 + d]  = accS[d] * iS;
#pragma unroll
            for (int d = 0; d < 10; ++d) out[O_NCOLOR + bk8 * 10 + d] = accC[d] * iC;
        }
        s_noninc[0 * NPAN + k] = nT;
        s_noninc[1 * NPAN + k] = nS;
        s_noninc[2 * NPAN + k] = nC;
    }
    __syncthreads();

    // rule probs: min over panels 0..7, one value per attribute per batch
    if (tid < 3) {
        float mn = s_noninc[tid * NPAN + 0];
#pragma unroll
        for (int kk = 1; kk < 8; ++kk) mn = fminf(mn, s_noninc[tid * NPAN + kk]);
        const int ro = (tid == 0) ? O_RTYPE : (tid == 1) ? O_RSIZE : O_RCOLOR;
        out[ro + b] = mn;
    }
}

extern "C" void kernel_launch(void* const* d_in, const int* in_sizes, int n_in,
                              void* d_out, int out_size, void* d_ws, size_t ws_size,
                              hipStream_t stream) {
    const float* ex = (const float*)d_in[0];   // (256,16,9,2)
    const float* ty = (const float*)d_in[1];   // (256,16,9,5)
    const float* sz = (const float*)d_in[2];   // (256,16,9,6)
    const float* co = (const float*)d_in[3];   // (256,16,9,10)
    float* out = (float*)d_out;

    const int B = in_sizes[0] / (NPAN * NSLOT * 2);   // 256
    scene_engine_kernel<<<B, 512, 0, stream>>>(ex, ty, sz, co, out);
}
